// StyledConv_25658134626759
// MI455X (gfx1250) — matrix operand order repaired and checked by reading, compiled, not run on hardware
//
#include <hip/hip_runtime.h>
#include <hip/hip_bf16.h>
#include <math.h>

// ---------------------------------------------------------------------------
// StyledConv (StyleGAN2): EqualLinear style -> modulated+demodulated 3x3 conv
// (batch-as-groups) -> FusedLeakyReLU.  Implicit GEMM on CDNA5 WMMA
// (v_wmma_f32_16x16x32_bf16), bf16 inputs / f32 accumulate.
// Block tile 128(M) x 128(N), 8 waves of 32x64, K staged 64 through LDS,
// double-buffered: A tile via GLOBAL_LOAD_ASYNC_TO_LDS_B128 (ASYNCcnt,
// partial-counter wait), B im2col gather prefetched into VGPRs.
// ---------------------------------------------------------------------------

#define B_SZ     8
#define IN_CH    512
#define OUT_CH   512
#define HW       1024          // 32*32
#define RED      (IN_CH * 9)   // 4608, reduction dim (tap-major: red = tap*512 + ic)
#define NST      (RED / 64)    // 72 K-stages

#define LIN_SCALE  0.044194173824159216f   // 1/sqrt(512)
#define CONV_SCALE 0.014731391274719738f   // 1/sqrt(512*9)
#define NEG_SLOPE  0.2f
#define ACT_SCALE  1.4142135623730951f

typedef __attribute__((ext_vector_type(16))) __bf16 bf16x16;
typedef __attribute__((ext_vector_type(8)))  __bf16 bf16x8;
typedef __attribute__((ext_vector_type(8)))  float  f32x8;
typedef __attribute__((ext_vector_type(4)))  int    v4i;

// LDS row stride in elements (64 data + 8 pad -> 144B rows, keeps 16B align)
#define LSTR 72
#define TILE_ELEMS (128 * LSTR)

#if defined(__HIP_DEVICE_COMPILE__) && \
    __has_builtin(__builtin_amdgcn_global_load_async_to_lds_b128)
#define USE_ASYNC_A 1
#else
#define USE_ASYNC_A 0
#endif

template <int N>
__device__ __forceinline__ void wait_async() {
#if USE_ASYNC_A
#if __has_builtin(__builtin_amdgcn_s_wait_asynccnt)
    __builtin_amdgcn_s_wait_asynccnt(N);
#else
    asm volatile("s_wait_asynccnt %0" :: "i"(N) : "memory");
#endif
#endif
}

__device__ __forceinline__ f32x8 wmma_bf16(bf16x16 a, bf16x16 b, f32x8 c) {
#if defined(__HIP_DEVICE_COMPILE__)
    return __builtin_amdgcn_wmma_f32_16x16x32_bf16(
        /*neg_a=*/false, a, /*neg_b=*/false, b,
        /*c_mod=*/(short)0, c, /*reuse_a=*/false, /*reuse_b=*/false);
#else
    (void)a; (void)b;
    return c;
#endif
}

// A fragment (16x32 bf16, MxK). ISA layout: lane<16 -> M=lane, K{0..7,16..23};
// lane>=16 -> M=lane-16, K{8..15,24..31}.  base = &tile[rowBase*LSTR + k0].
__device__ __forceinline__ bf16x16 load_fragA(const __bf16* base, int lane) {
    const __bf16* p = base + (lane & 15) * LSTR + ((lane >> 4) << 3);
    bf16x8 lo = *(const bf16x8*)(p);
    bf16x8 hi = *(const bf16x8*)(p + 16);
    return __builtin_shufflevector(lo, hi, 0,1,2,3,4,5,6,7,8,9,10,11,12,13,14,15);
}

// B fragment (32x16 bf16, KxN), tile stored transposed as [n][k].
// ISA layout: lane<16 -> N=lane, K{0..7,16..23}; lane>=16 -> N=lane-16, K{8..15,24..31}.
__device__ __forceinline__ bf16x16 load_fragB(const __bf16* base, int lane) {
    const __bf16* p = base + (lane & 15) * LSTR + ((lane >> 4) << 3);
    bf16x8 lo = *(const bf16x8*)(p);
    bf16x8 hi = *(const bf16x8*)(p + 16);
    return __builtin_shufflevector(lo, hi, 0,1,2,3,4,5,6,7,8,9,10,11,12,13,14,15);
}

// Issue one K-stage of the A tile (128 x 64 bf16) into an LDS tile.
__device__ __forceinline__ void stage_a(const __bf16* wB, __bf16* dstTile,
                                        int kBase, int tid) {
    #pragma unroll
    for (int i = 0; i < 4; ++i) {
        int c   = tid + i * 256;       // 1024 chunks of 8 elems
        int row = c >> 3;              // 0..127
        int col = (c & 7) * 8;         // 0..56
#if USE_ASYNC_A
        __builtin_amdgcn_global_load_async_to_lds_b128(
            (__attribute__((address_space(1))) v4i*)
                (wB + (size_t)row * RED + kBase + col),
            (__attribute__((address_space(3))) v4i*)
                (dstTile + row * LSTR + col),
            0, 0);
#else
        uint4 v = *(const uint4*)(wB + (size_t)row * RED + kBase + col);
        *(uint4*)(dstTile + row * LSTR + col) = v;
#endif
    }
}

// Gather one K-stage of B (im2col) into registers: 32 consecutive hw (one
// aligned image row) for one input channel.
__device__ __forceinline__ void gather_b(float* regs, const float* inB,
                                         int kBase, int nBase, int kk, int n0) {
    const int tap = kBase >> 9;        // 0..8 (stage never straddles a tap)
    const int dy  = tap / 3 - 1;
    const int dx  = tap % 3 - 1;
    const int yy  = ((nBase + n0) >> 5) + dy;   // constant per thread
    const bool rowOk = (unsigned)yy < 32u;
    const float* srcRow =
        inB + (size_t)((kBase & 511) + kk) * HW + (yy << 5) + dx;
    #pragma unroll
    for (int nn = 0; nn < 32; ++nn) {
        int x = nn + dx;
        regs[nn] = (rowOk && (unsigned)x < 32u) ? srcRow[nn] : 0.f;
    }
}

// ---------------------------------------------------------------------------
// Kernel 1: s[b, ic] = style[b,:] . mod_w[ic,:] * LIN_SCALE + mod_b[ic]
// ---------------------------------------------------------------------------
__global__ __launch_bounds__(256) void style_kernel(
    const float* __restrict__ style, const float* __restrict__ mod_w,
    const float* __restrict__ mod_b, float* __restrict__ s)
{
    int g  = blockIdx.x * 256 + threadIdx.x;   // 0..4095
    int b  = g >> 9;
    int ic = g & 511;
    const float* st = style + b * 512;
    const float* mw = mod_w + (size_t)ic * 512;
    float acc = 0.f;
    #pragma unroll 8
    for (int d = 0; d < 512; ++d) acc += st[d] * mw[d];
    s[g] = acc * LIN_SCALE + mod_b[ic];
}

// ---------------------------------------------------------------------------
// Kernel 2: per (b, oc): modulate, demodulate, emit bf16 weight in tap-major
// layout  wmod[((b*OUT_CH + oc) * RED) + tap*512 + ic]
// ---------------------------------------------------------------------------
__global__ __launch_bounds__(256) void modw_kernel(
    const float* __restrict__ weight, const float* __restrict__ s,
    __bf16* __restrict__ wmod)
{
    __shared__ float red[256];
    const int b   = blockIdx.y;
    const int oc  = blockIdx.x;
    const int tid = threadIdx.x;
    const float* w  = weight + (size_t)oc * RED;    // [ic][ky][kx] flat
    const float* sb = s + b * IN_CH;

    float part = 0.f;
    for (int idx = tid; idx < RED; idx += 256) {
        int ic = idx / 9;
        float v = CONV_SCALE * w[idx] * sb[ic];
        part += v * v;
    }
    red[tid] = part;
    __syncthreads();
    #pragma unroll
    for (int off = 128; off > 0; off >>= 1) {
        if (tid < off) red[tid] += red[tid + off];
        __syncthreads();
    }
    const float demod = rsqrtf(red[0] + 1e-8f);

    __bf16* wo = wmod + (size_t)(b * OUT_CH + oc) * RED;
    for (int idx = tid; idx < RED; idx += 256) {
        int ic  = idx / 9;
        int tap = idx - ic * 9;
        float v = CONV_SCALE * w[idx] * sb[ic] * demod;
        wo[tap * 512 + ic] = (__bf16)v;
    }
}

// ---------------------------------------------------------------------------
// Kernel 3: implicit-GEMM conv + bias + leaky_relu*sqrt(2), double-buffered.
// Per batch:  D[oc, hw] = sum_red  W[oc, red] * X[red, hw]
// ---------------------------------------------------------------------------
__global__ __launch_bounds__(256) void conv_kernel(
    const float* __restrict__ in, const __bf16* __restrict__ wmod,
    const float* __restrict__ bias, float* __restrict__ out)
{
    __shared__ __bf16 aT[2][TILE_ELEMS];  // A tiles: [m][k] 128 x 64 (+pad)
    __shared__ __bf16 bT[2][TILE_ELEMS];  // B tiles: [n][k] 128 x 64 (+pad)

    const int b      = blockIdx.z;
    const int mBlock = blockIdx.y * 128;
    const int nBase  = blockIdx.x * 128;
    const int tid    = threadIdx.x;
    const int lane   = tid & 31;
    const int wid    = tid >> 5;
    const int waveM  = (wid & 3) * 32;    // 0,32,64,96
    const int waveN  = (wid >> 2) * 64;   // 0,64

    f32x8 acc[2][4] = {};

    const float*  inB = in   + (size_t)b * IN_CH * HW;
    const __bf16* wB  = wmod + (size_t)(b * OUT_CH + mBlock) * RED;

    // B-gather coordinates (fixed per thread)
    const int kk = tid >> 2;             // 0..63 -> ic = (kBase&511) + kk
    const int n0 = (tid & 3) * 32;       // one aligned image row

    float regs[32];

    // ---- prologue: stage 0 in flight ----
    gather_b(regs, inB, 0, nBase, kk, n0);
    stage_a(wB, aT[0], 0, tid);

    for (int s = 0; s < NST; ++s) {
        const int cur = s & 1;
        // store pre-gathered B(s) into LDS (transposed [n][k])
        #pragma unroll
        for (int nn = 0; nn < 32; ++nn)
            bT[cur][(n0 + nn) * LSTR + kk] = (__bf16)regs[nn];

        if (s + 1 < NST) {
            const int kNext = (s + 1) * 64;
            stage_a(wB, aT[cur ^ 1], kNext, tid);     // async, in flight
            gather_b(regs, inB, kNext, nBase, kk, n0); // loads in flight
            wait_async<4>();   // 4/thread belong to s+1; in-order => s done
        } else {
            wait_async<0>();
        }
        __syncthreads();

        // ---- compute stage s: 2 K-steps of 32, 8 WMMAs each ----
        const __bf16* aB = aT[cur];
        const __bf16* bB = bT[cur];
        #pragma unroll
        for (int ks = 0; ks < 2; ++ks) {
            const int k0 = ks * 32;
            bf16x16 a0  = load_fragA(aB + (waveM     ) * LSTR + k0, lane);
            bf16x16 a1  = load_fragA(aB + (waveM + 16) * LSTR + k0, lane);
            bf16x16 bf0 = load_fragB(bB + (waveN     ) * LSTR + k0, lane);
            bf16x16 bf1 = load_fragB(bB + (waveN + 16) * LSTR + k0, lane);
            bf16x16 bf2 = load_fragB(bB + (waveN + 32) * LSTR + k0, lane);
            bf16x16 bf3 = load_fragB(bB + (waveN + 48) * LSTR + k0, lane);
            acc[0][0] = wmma_bf16(a0, bf0, acc[0][0]);
            acc[1][0] = wmma_bf16(a1, bf0, acc[1][0]);
            acc[0][1] = wmma_bf16(a0, bf1, acc[0][1]);
            acc[1][1] = wmma_bf16(a1, bf1, acc[1][1]);
            acc[0][2] = wmma_bf16(a0, bf2, acc[0][2]);
            acc[1][2] = wmma_bf16(a1, bf2, acc[1][2]);
            acc[0][3] = wmma_bf16(a0, bf3, acc[0][3]);
            acc[1][3] = wmma_bf16(a1, bf3, acc[1][3]);
        }
        __syncthreads();   // seal buffer reuse for stage s+1 writers
    }

    // ---- epilogue: bias + leaky_relu(0.2) * sqrt(2) ----
    // C/D layout: VGPR r -> M = r (lanes 0-15) or r+8 (lanes 16-31); N = lane&15
    const int l15   = lane & 15;
    const int mLane = (lane >> 4) << 3;
    #pragma unroll
    for (int mf = 0; mf < 2; ++mf) {
        #pragma unroll
        for (int nf = 0; nf < 4; ++nf) {
            int hw = nBase + waveN + nf * 16 + l15;
            #pragma unroll
            for (int r = 0; r < 8; ++r) {
                int oc  = mBlock + waveM + mf * 16 + mLane + r;
                float v = acc[mf][nf][r] + bias[oc];
                v = (v > 0.f ? v : v * NEG_SLOPE) * ACT_SCALE;
                out[(((size_t)b * OUT_CH + oc) << 10) + hw] = v;
            }
        }
    }
}

// ---------------------------------------------------------------------------
// Host entry
// ---------------------------------------------------------------------------
extern "C" void kernel_launch(void* const* d_in, const int* in_sizes, int n_in,
                              void* d_out, int out_size, void* d_ws, size_t ws_size,
                              hipStream_t stream) {
    const float* input    = (const float*)d_in[0];  // [8,512,32,32]
    const float* style    = (const float*)d_in[1];  // [8,512]
    const float* weight   = (const float*)d_in[2];  // [512,512,3,3]
    const float* mod_w    = (const float*)d_in[3];  // [512,512]
    const float* mod_b    = (const float*)d_in[4];  // [512]
    const float* flr_bias = (const float*)d_in[5];  // [512]
    float* out = (float*)d_out;                     // [8,512,32,32]

    // workspace: s (8*512 f32 = 16KB) then bf16 wmod (8*512*4608*2 = 37.75MB)
    float*  s    = (float*)d_ws;
    __bf16* wmod = (__bf16*)((char*)d_ws + (size_t)B_SZ * IN_CH * sizeof(float));

    style_kernel<<<16, 256, 0, stream>>>(style, mod_w, mod_b, s);
    modw_kernel<<<dim3(OUT_CH, B_SZ), 256, 0, stream>>>(weight, s, wmod);
    conv_kernel<<<dim3(HW / 128, OUT_CH / 128, B_SZ), 256, 0, stream>>>(
        input, wmod, flr_bias, out);
}
